// GIN_1752346657103
// MI455X (gfx1250) — matrix operand order repaired and checked by reading, not compiled, not run on hardware
//
#include <hip/hip_runtime.h>
#include <hip/hip_bf16.h>
#include <math.h>

typedef __attribute__((ext_vector_type(16))) _Float16 v16h;
typedef __attribute__((ext_vector_type(8)))  float    v8f;

#define NFEAT 64
#define NHID  128

// ---------------------------------------------------------------------------
// init: agg = x (GIN: h = (1+eps)*x + sum_neigh, eps=0), zero BN stat slots
// ---------------------------------------------------------------------------
__global__ void init_kernel(const float* __restrict__ x, float* __restrict__ agg,
                            float* __restrict__ stats, int n4) {
  int i = blockIdx.x * blockDim.x + threadIdx.x;
  if (i < n4) ((float4*)agg)[i] = ((const float4*)x)[i];
  if (i < 256) stats[i] = 0.0f;
}

// ---------------------------------------------------------------------------
// scatter: agg[dst] += x[src]; one lane per (edge, feature-pair)
// ---------------------------------------------------------------------------
__global__ void scatter_kernel(const float* __restrict__ x, const int* __restrict__ ei,
                               float* __restrict__ agg, int E) {
  unsigned tid = blockIdx.x * blockDim.x + threadIdx.x;
  unsigned e = tid >> 5;
  if (e >= (unsigned)E) return;
  int f = (int)(tid & 31u) * 2;
  int s = ei[e];
  int d = ei[(unsigned)E + e];
  float2 v = *(const float2*)(x + (size_t)s * NFEAT + f);
  atomicAdd(agg + (size_t)d * NFEAT + f,     v.x);
  atomicAdd(agg + (size_t)d * NFEAT + f + 1, v.y);
}

// ---------------------------------------------------------------------------
// spectral-norm (1 power iteration, torch semantics) + pack W/sigma into
// WMMA B-fragment layout: frag[((tile*ksteps+kk)*32+lane)*16 + e]
//   lane 0-15 : column n=tile*16+lane,   K = kk*32 + (e<8 ? e : 8+e)
//   lane 16-31: column n=tile*16+lane-16,K = kk*32 + 8 + (e<8 ? e : 8+e)
// B[k][n] = W_sn[n][k]  (GEMM computes h @ W_sn^T)
// ---------------------------------------------------------------------------
__global__ __launch_bounds__(128)
void sn_pack_kernel(const float* __restrict__ W, const float* __restrict__ u,
                    int nOut, int nIn, _Float16* __restrict__ frag) {
  __shared__ float red[128];
  __shared__ float vsh[128];
  __shared__ float sInv;
  int t = threadIdx.x;

  // v = l2norm(W^T u)
  float vk = 0.0f;
  if (t < nIn)
    for (int n = 0; n < nOut; ++n) vk += W[n * nIn + t] * u[n];
  red[t] = (t < nIn) ? vk * vk : 0.0f;
  __syncthreads();
  for (int s = 64; s > 0; s >>= 1) {
    if (t < s) red[t] += red[t + s];
    __syncthreads();
  }
  float vnorm = sqrtf(red[0]) + 1e-12f;
  vsh[t] = (t < nIn) ? vk / vnorm : 0.0f;
  __syncthreads();

  // wv = W v ; sigma = u_new . (W v) = ||wv||^2 / (||wv|| + eps)
  float wv = 0.0f;
  if (t < nOut)
    for (int k = 0; k < nIn; ++k) wv += W[t * nIn + k] * vsh[k];
  red[t] = (t < nOut) ? wv * wv : 0.0f;
  __syncthreads();
  for (int s = 64; s > 0; s >>= 1) {
    if (t < s) red[t] += red[t + s];
    __syncthreads();
  }
  if (t == 0) {
    float n2 = red[0];
    float sigma = n2 / (sqrtf(n2) + 1e-12f);
    sInv = 1.0f / sigma;
  }
  __syncthreads();
  float inv = sInv;

  int ksteps = nIn >> 5;
  int ntiles = nOut >> 4;
  int total = ntiles * ksteps * 32 * 16;
  for (int i = t; i < total; i += 128) {
    int e    = i & 15;
    int lane = (i >> 4) & 31;
    int rem  = i >> 9;
    int kk   = rem % ksteps;
    int tile = rem / ksteps;
    int col  = tile * 16 + (lane & 15);
    int k    = kk * 32 + ((lane >= 16) ? 8 : 0) + ((e < 8) ? e : 8 + e);
    frag[i] = (_Float16)(W[col * nIn + k] * inv);
  }
}

// ---------------------------------------------------------------------------
// GEMM1: h = relu((x+agg) @ W1sn^T + b1)  (f16 out) + BN sum/sumsq stats
// one wave per 16-row strip, N=128 as 8 WMMA tiles, K=64 as 2 k-steps
// ---------------------------------------------------------------------------
__global__ __launch_bounds__(256)
void gemm1_kernel(const float* __restrict__ agg, const _Float16* __restrict__ bfrag,
                  const float* __restrict__ b1, _Float16* __restrict__ h,
                  float* __restrict__ bnsum, float* __restrict__ bnsq, int nstrips) {
  __shared__ float sSum[NHID];
  __shared__ float sSq[NHID];
  int t = threadIdx.x;
  if (t < NHID) { sSum[t] = 0.0f; sSq[t] = 0.0f; }
  __syncthreads();

  int wave = t >> 5, lane = t & 31;
  int strip = blockIdx.x * 8 + wave;
  if (strip < nstrips) {
    int row   = strip * 16 + (lane & 15);
    int khalf = (lane >= 16) ? 8 : 0;
    const float* rp = agg + (size_t)row * NFEAT;

    v16h a[2];
    #pragma unroll
    for (int kk = 0; kk < 2; ++kk) {
      int kb = kk * 32 + khalf;
      #pragma unroll
      for (int e = 0; e < 8; ++e) a[kk][e]     = (_Float16)rp[kb + e];
      #pragma unroll
      for (int e = 0; e < 8; ++e) a[kk][8 + e] = (_Float16)rp[kb + 16 + e];
    }

    int nlo    = lane & 15;
    int rowout = strip * 16 + ((lane >= 16) ? 8 : 0);
    #pragma unroll
    for (int tile = 0; tile < 8; ++tile) {
      v8f c = {};
      const v16h* bp = (const v16h*)bfrag + (size_t)(tile * 2) * 32;
      v16h bv0 = bp[lane];
      v16h bv1 = bp[32 + lane];
      c = __builtin_amdgcn_wmma_f32_16x16x32_f16(false, a[0], false, bv0, (short)0, c, false, false);
      c = __builtin_amdgcn_wmma_f32_16x16x32_f16(false, a[1], false, bv1, (short)0, c, false, false);

      int n = tile * 16 + nlo;
      float bias = b1[n];
      float s = 0.0f, s2 = 0.0f;
      #pragma unroll
      for (int r = 0; r < 8; ++r) {
        float vv = c[r] + bias;
        vv = vv > 0.0f ? vv : 0.0f;
        s += vv; s2 += vv * vv;
        h[(size_t)(rowout + r) * NHID + n] = (_Float16)vv;
      }
      // lanes l and l+16 hold the same channel n for different rows
      s  += __shfl_down(s, 16, 32);
      s2 += __shfl_down(s2, 16, 32);
      if (lane < 16) {
        atomicAdd(&sSum[n], s);
        atomicAdd(&sSq[n], s2);
      }
    }
  }
  __syncthreads();
  if (t < NHID) {
    atomicAdd(&bnsum[t], sSum[t]);
    atomicAdd(&bnsq[t],  sSq[t]);
  }
}

// ---------------------------------------------------------------------------
// BN finalize: fold mean/var/gamma/beta into per-channel scale/shift
// ---------------------------------------------------------------------------
__global__ void bn_finalize_kernel(const float* __restrict__ stats,
                                   const float* __restrict__ gamma,
                                   const float* __restrict__ beta,
                                   float* __restrict__ scsh, float invN) {
  int t = threadIdx.x;
  float mean = stats[t] * invN;
  float var  = stats[NHID + t] * invN - mean * mean;
  float sc   = gamma[t] * rsqrtf(var + 1e-5f);
  scsh[t]        = sc;
  scsh[NHID + t] = beta[t] - mean * sc;
}

// ---------------------------------------------------------------------------
// GEMM2: out = (BN(h)) @ W2sn^T + b2   (f32 out), K=128 as 4 k-steps
// ---------------------------------------------------------------------------
__global__ __launch_bounds__(256)
void gemm2_kernel(const _Float16* __restrict__ h, const _Float16* __restrict__ bfrag,
                  const float* __restrict__ b2, const float* __restrict__ scsh,
                  float* __restrict__ out, int nstrips) {
  __shared__ float sScale[NHID];
  __shared__ float sShift[NHID];
  int t = threadIdx.x;
  if (t < NHID) { sScale[t] = scsh[t]; sShift[t] = scsh[NHID + t]; }
  __syncthreads();

  int wave = t >> 5, lane = t & 31;
  int strip = blockIdx.x * 8 + wave;
  if (strip >= nstrips) return;

  int row   = strip * 16 + (lane & 15);
  int khalf = (lane >= 16) ? 8 : 0;
  const _Float16* rp = h + (size_t)row * NHID;

  v16h a[4];
  #pragma unroll
  for (int kk = 0; kk < 4; ++kk) {
    int kb = kk * 32 + khalf;
    #pragma unroll
    for (int e = 0; e < 8; ++e) {
      int k = kb + e;
      a[kk][e] = (_Float16)((float)rp[k] * sScale[k] + sShift[k]);
    }
    #pragma unroll
    for (int e = 0; e < 8; ++e) {
      int k = kb + 16 + e;
      a[kk][8 + e] = (_Float16)((float)rp[k] * sScale[k] + sShift[k]);
    }
  }

  int nlo    = lane & 15;
  int rowout = strip * 16 + ((lane >= 16) ? 8 : 0);
  #pragma unroll
  for (int tile = 0; tile < 8; ++tile) {
    v8f c = {};
    const v16h* bp = (const v16h*)bfrag + (size_t)(tile * 4) * 32;
    #pragma unroll
    for (int kk = 0; kk < 4; ++kk) {
      v16h bv = bp[kk * 32 + lane];
      c = __builtin_amdgcn_wmma_f32_16x16x32_f16(false, a[kk], false, bv, (short)0, c, false, false);
    }
    int n = tile * 16 + nlo;
    float bias = b2[n];
    #pragma unroll
    for (int r = 0; r < 8; ++r)
      out[(size_t)(rowout + r) * NHID + n] = c[r] + bias;
  }
}

// ---------------------------------------------------------------------------
extern "C" void kernel_launch(void* const* d_in, const int* in_sizes, int n_in,
                              void* d_out, int out_size, void* d_ws, size_t ws_size,
                              hipStream_t stream) {
  const float* x     = (const float*)d_in[0];
  const int*   ei    = (const int*)d_in[1];
  const float* W1    = (const float*)d_in[2];
  const float* b1    = (const float*)d_in[3];
  const float* u1    = (const float*)d_in[4];
  const float* gamma = (const float*)d_in[5];
  const float* beta  = (const float*)d_in[6];
  const float* W2    = (const float*)d_in[7];
  const float* b2    = (const float*)d_in[8];
  const float* u2    = (const float*)d_in[9];
  float* out = (float*)d_out;

  int N = in_sizes[0] / NFEAT;   // 65536
  int E = in_sizes[1] / 2;       // 1048576

  // workspace layout
  char* ws = (char*)d_ws;
  float*     agg   = (float*)ws;                                   // N*64*4  = 16 MB
  _Float16*  hbuf  = (_Float16*)(ws + ((size_t)N * NFEAT * 4));    // N*128*2 = 16 MB
  char*      tail  = ws + (size_t)N * NFEAT * 4 + (size_t)N * NHID * 2;
  _Float16*  bf1   = (_Float16*)tail;                              // 8*2*32*16 halves = 16 KB
  _Float16*  bf2   = (_Float16*)(tail + (64 << 10));               // 8*4*32*16 halves = 32 KB
  float*     stats = (float*)(tail + (128 << 10));                 // 256 floats
  float*     scsh  = (float*)(tail + (132 << 10));                 // 256 floats

  // 1) agg = x, zero BN stats
  int n4 = N * NFEAT / 4;
  init_kernel<<<(n4 + 255) / 256, 256, 0, stream>>>(x, agg, stats, n4);

  // 2) spectral-norm + B-fragment packing (tiny)
  sn_pack_kernel<<<1, 128, 0, stream>>>(W1, u1, NHID, NFEAT, bf1);
  sn_pack_kernel<<<1, 128, 0, stream>>>(W2, u2, NHID, NHID,  bf2);

  // 3) neighbor-sum scatter (L2-resident atomics)
  unsigned sthreads = (unsigned)E * 32u;
  scatter_kernel<<<(sthreads + 255u) / 256u, 256, 0, stream>>>(x, ei, agg, E);

  // 4) GEMM1 + bias + ReLU + BN stats
  int nstrips = N / 16;
  int gblocks = (nstrips + 7) / 8;
  gemm1_kernel<<<gblocks, 256, 0, stream>>>(agg, bf1, b1, hbuf, stats, stats + NHID, nstrips);

  // 5) fold BN stats into scale/shift
  bn_finalize_kernel<<<1, NHID, 0, stream>>>(stats, gamma, beta, scsh, 1.0f / (float)N);

  // 6) GEMM2 with fused BN affine + bias
  gemm2_kernel<<<gblocks, 256, 0, stream>>>(hbuf, bf2, b2, scsh, out, nstrips);
}
